// Spatial_layer_52295521796477
// MI455X (gfx1250) — matrix operand order repaired and checked by reading, not compiled, not run on hardware
//
#include <hip/hip_runtime.h>
#include <hip/hip_bf16.h>

// ---------------------------------------------------------------------------
// MI455X (gfx1250) implementation, wave32 + V_WMMA_F32_16X16X32_BF16.
// Round 2: register-blocked GEMMs (16x64 strip per wave, A-fragment reuse x4),
// hoisted q fragments in attention, prefetch on long-K GEMM.
// Pipeline (all on `stream`, scratch in d_ws):
//   k_prep  : f32->bf16 weight conversion (in_proj_w, transposed theta)
//   k_pool  : SE adaptive-avg-pool on raw-reshaped view          [B,C]
//   k_fc    : SE fc1(relu) / fc2(sigmoid)                        [B,C]
//   k_conv  : SE-scale + 1x1 conv over L -> x_in f32/bf16 + xg   [B,E,C],[B,C,E]
//   k_qkv   : WMMA GEMM xg @ W_qkv^T + bias (q scaled by 1/8)    bf16 q,k
//   k_attn  : WMMA scores + LDS softmax + head-mean -> b_adj     (d_out tail)
//   k_adjxg : WMMA GEMM b_adj @ xg -> bf16                       [B,C,E]
//   k_cheb  : WMMA GEMM xg@theta0 + adjxg@theta1 -> f32          [B,C,E]
//   k_epi   : out = relu(x_in + relu(cheb)) (flat identity)      (d_out head)
// ---------------------------------------------------------------------------

#define Bn  32
#define Ln  128
#define Cn  1024
#define En  256

typedef __bf16 bf16;
typedef __attribute__((ext_vector_type(16))) __bf16 v16bf;
typedef __attribute__((ext_vector_type(8)))  float  v8f;

// ---- WMMA fragment builders (layouts per CDNA5 ISA §7.12.2) ----

// A 16x32 bf16: lane L: m=L&15, half=L>>4; elem j -> k = 8*half + (j<8 ? j : j+8)
static __device__ __forceinline__ v16bf frag_a_bf(const bf16* base, int ld) {
  const int lane = threadIdx.x & 31;
  const bf16* row = base + (lane & 15) * ld + 8 * (lane >> 4);
  v16bf f;
#pragma unroll
  for (int j = 0; j < 16; ++j) {
    const int k = (j < 8) ? j : (j + 8);
    f[j] = row[k];
  }
  return f;
}

// Same, converting from f32 source (used for b_adj which lives in d_out as f32)
static __device__ __forceinline__ v16bf frag_a_f32(const float* base, int ld) {
  const int lane = threadIdx.x & 31;
  const float* row = base + (lane & 15) * ld + 8 * (lane >> 4);
  v16bf f;
#pragma unroll
  for (int j = 0; j < 16; ++j) {
    const int k = (j < 8) ? j : (j + 8);
    f[j] = (bf16)row[k];
  }
  return f;
}

// B 32x16 bf16: lane L: n=L&15, half=L>>4; elem j -> k = 8*half + (j<8 ? j : j+8).
// Caller provides base s.t. element (k,n) lives at base[n*ld + k]  (i.e. the
// source is stored N-major so every fragment read is contiguous in memory).
static __device__ __forceinline__ v16bf frag_b_bf(const bf16* base, int ld) {
  const int lane = threadIdx.x & 31;
  const bf16* col = base + (lane & 15) * ld + 8 * (lane >> 4);
  v16bf f;
#pragma unroll
  for (int j = 0; j < 16; ++j) f[j] = col[(j < 8) ? j : (j + 8)];
  return f;
}

static __device__ __forceinline__ v8f wmma_bf16(v16bf a, v16bf b, v8f c) {
  return __builtin_amdgcn_wmma_f32_16x16x32_bf16(
      /*neg_a=*/false, a, /*neg_b=*/false, b,
      /*c_mod=*/(short)0, c, /*reuse_a=*/false, /*reuse_b=*/false);
}

// ---------------------------------------------------------------------------
// k_prep: bf16 weight staging. wqkv = bf16(in_proj_w) [3E,E] row-major;
// thetaT[t][n][k] = theta[t][k][n]  (so B-fragment reads are contiguous).
__global__ void k_prep(const float* __restrict__ wqkv_f,
                       const float* __restrict__ theta_f,
                       bf16* __restrict__ wqkv, bf16* __restrict__ thetaT) {
  const int i = blockIdx.x * blockDim.x + threadIdx.x;
  if (i < 3 * En * En) wqkv[i] = (bf16)wqkv_f[i];
  const int j = i - 3 * En * En;
  if (j >= 0 && j < 2 * En * En) {
    const int t = j >> 16, n = (j >> 8) & 255, k = j & 255;
    thetaT[j] = (bf16)theta_f[t * 65536 + k * 256 + n];
  }
}

// k_pool: y0[b,c'] = mean_l x_flat[b, c'*128 + l]   (raw-reshape view)
__global__ void k_pool(const float* __restrict__ x, float* __restrict__ y0) {
  const int i = blockIdx.x * blockDim.x + threadIdx.x;   // 32768
  const int b = i >> 10, c = i & 1023;
  const float* p = x + b * (Ln * Cn) + c * Ln;
  float s = 0.f;
#pragma unroll 4
  for (int l = 0; l < Ln; ++l) s += p[l];
  y0[i] = s * (1.0f / Ln);
}

// k_fc: out[b,j] = act(sum_k in[b,k] * W[j,k]); act 0=relu, 1=sigmoid
__global__ void k_fc(const float* __restrict__ in, const float* __restrict__ W,
                     float* __restrict__ out, int act) {
  const int i = blockIdx.x * blockDim.x + threadIdx.x;   // 32768
  const int b = i >> 10, j = i & 1023;
  const float* a = in + b * Cn;
  const float* w = W + j * Cn;
  float s = 0.f;
#pragma unroll 4
  for (int k = 0; k < Cn; ++k) s += a[k] * w[k];
  out[i] = (act == 0) ? fmaxf(s, 0.f) : (1.0f / (1.0f + __expf(-s)));
}

// k_conv: x_in[b,e,c] = sum_l W[e,l] * x[b,l,c] * y2[b, l*8 + (c>>7)]
// Writes f32 x_in [B,E,C], bf16 xinb [B,E,C], bf16 xg [B,C,E].
__global__ void k_conv(const float* __restrict__ x, const float* __restrict__ wsl,
                       const float* __restrict__ y2, float* __restrict__ x_in,
                       bf16* __restrict__ xinb, bf16* __restrict__ xg) {
  const int i = blockIdx.x * blockDim.x + threadIdx.x;   // 8388608
  const int b = i >> 18, r = i & 262143, e = r >> 10, c = r & 1023;
  const float* xb = x + b * (Ln * Cn) + c;
  const float* yb = y2 + b * Cn + (c >> 7);
  const float* we = wsl + e * Ln;
  float s = 0.f;
#pragma unroll 4
  for (int l = 0; l < Ln; ++l) s += we[l] * xb[l * Cn] * yb[l * 8];
  x_in[i] = s;
  xinb[i] = (bf16)s;
  xg[b * 262144 + c * 256 + e] = (bf16)s;
}

// k_qkv: qkv[:, :512] = xg @ W^T + bias; q scaled by 1/sqrt(hd)=0.125 at store.
// Each wave: one 16x64 strip (4 accumulators), A fragment reused 4x per k-step.
__global__ __launch_bounds__(128) void k_qkv(const bf16* __restrict__ xg,
                                             const bf16* __restrict__ wqkv,
                                             const float* __restrict__ bias,
                                             bf16* __restrict__ qb, bf16* __restrict__ kb) {
  const int wid = blockIdx.x * 4 + (threadIdx.x >> 5);   // 16384 waves
  const int b = wid >> 9, r = wid & 511;
  const int m0 = (r >> 3) * 16;        // 64 m-tiles
  const int ns = (r & 7) * 64;         // 8 n-strips over qkv cols 0..511
  const bf16* A = xg + b * 262144 + m0 * 256;
  v8f acc[4] = {};
#pragma unroll
  for (int ks = 0; ks < 8; ++ks) {
    const v16bf a = frag_a_bf(A + ks * 32, 256);
#pragma unroll
    for (int t = 0; t < 4; ++t)
      acc[t] = wmma_bf16(a, frag_b_bf(wqkv + (ns + t * 16) * 256 + ks * 32, 256), acc[t]);
  }
  const int lane = threadIdx.x & 31, n = lane & 15, half = lane >> 4;
#pragma unroll
  for (int t = 0; t < 4; ++t) {
    const int col = ns + t * 16 + n;
    const float bv = bias[col];
#pragma unroll
    for (int i = 0; i < 8; ++i) {
      const int m = m0 + i + 8 * half;
      const float v = acc[t][i] + bv;
      if (col < 256) qb[b * 262144 + m * 256 + col] = (bf16)(v * 0.125f);
      else           kb[b * 262144 + m * 256 + (col - 256)] = (bf16)v;
    }
  }
}

// k_attn: per (b, 16-row q tile): 4 heads of WMMA scores -> LDS softmax ->
// head-mean accumulated in LDS -> streamed to b_adj slot of d_out.
// q fragments hoisted out of the column loop (reused 16x per head).
__global__ __launch_bounds__(128) void k_attn(const bf16* __restrict__ qb,
                                              const bf16* __restrict__ kb,
                                              float* __restrict__ badj_out) {
  extern __shared__ float sm[];
  float* sc = sm;             // 16*1024 score strip
  float* ac = sm + 16384;     // 16*1024 head-mean accumulator
  float* st = sm + 32768;     // 160 floats of reduction scratch
  const int b = blockIdx.x >> 6, m0 = (blockIdx.x & 63) * 16;
  const int tid = threadIdx.x, wv = tid >> 5, lane = tid & 31;
  const int fn = lane & 15, fh = lane >> 4;
  const bf16* qbb = qb + b * 262144 + m0 * 256;
  const bf16* kbb = kb + b * 262144;
  for (int i = tid; i < 16384; i += 128) ac[i] = 0.0f;
  __syncthreads();
  for (int h = 0; h < 4; ++h) {
    const v16bf qa0 = frag_a_bf(qbb + h * 64,      256);
    const v16bf qa1 = frag_a_bf(qbb + h * 64 + 32, 256);
    for (int nt = wv; nt < 64; nt += 4) {
      v8f acc = {};
      acc = wmma_bf16(qa0, frag_b_bf(kbb + nt * 16 * 256 + h * 64,      256), acc);
      acc = wmma_bf16(qa1, frag_b_bf(kbb + nt * 16 * 256 + h * 64 + 32, 256), acc);
#pragma unroll
      for (int i = 0; i < 8; ++i) sc[(i + 8 * fh) * 1024 + nt * 16 + fn] = acc[i];
    }
    __syncthreads();
    const int row = tid >> 3, seg = tid & 7;
    float* rp = sc + row * 1024 + seg * 128;
    float mx = -3.4e38f;
    for (int j = 0; j < 128; ++j) mx = fmaxf(mx, rp[j]);
    st[row * 8 + seg] = mx;
    __syncthreads();
    if (seg == 0) {
      float m2 = st[row * 8];
      for (int j = 1; j < 8; ++j) m2 = fmaxf(m2, st[row * 8 + j]);
      st[128 + row] = m2;
    }
    __syncthreads();
    const float rm = st[128 + row];
    float s = 0.0f;
    for (int j = 0; j < 128; ++j) { const float e = __expf(rp[j] - rm); rp[j] = e; s += e; }
    st[row * 8 + seg] = s;
    __syncthreads();
    if (seg == 0) {
      float s2 = 0.f;
      for (int j = 0; j < 8; ++j) s2 += st[row * 8 + j];
      st[144 + row] = s2;
    }
    __syncthreads();
    const float inv = 0.25f / st[144 + row];   // mean over 4 heads folded in
    float* ap = ac + row * 1024 + seg * 128;
    for (int j = 0; j < 128; ++j) ap[j] += rp[j] * inv;
    __syncthreads();
  }
  float* po = badj_out + b * 1048576 + m0 * 1024;
  for (int i = tid; i < 16384; i += 128) po[i] = ac[i];
}

// k_adjxg: adjxg = b_adj @ xg  (A f32->bf16 on the fly; B via xinb = xg^T).
// Each wave: one 16x64 strip (4 accumulators), K=1024, prefetch next A strip.
__global__ __launch_bounds__(128) void k_adjxg(const float* __restrict__ badj,
                                               const bf16* __restrict__ xinb,
                                               bf16* __restrict__ adjxg) {
  const int wid = blockIdx.x * 4 + (threadIdx.x >> 5);   // 8192 waves
  const int b = wid >> 8, r = wid & 255;
  const int m0 = (r >> 2) * 16;        // 64 m-tiles
  const int ns = (r & 3) * 64;         // 4 n-strips over E=256
  const float* A  = badj + b * 1048576 + m0 * 1024;
  const bf16*  Bb = xinb + b * 262144;
  v8f acc[4] = {};
  for (int ks = 0; ks < 32; ++ks) {
    __builtin_prefetch(A + (ks + 4) * 32, 0, 0);         // global_prefetch_b8
    const v16bf a = frag_a_f32(A + ks * 32, 1024);
#pragma unroll
    for (int t = 0; t < 4; ++t)
      acc[t] = wmma_bf16(a, frag_b_bf(Bb + (ns + t * 16) * 1024 + ks * 32, 1024), acc[t]);
  }
  const int lane = threadIdx.x & 31, n = lane & 15, half = lane >> 4;
#pragma unroll
  for (int t = 0; t < 4; ++t) {
    bf16* o = adjxg + b * 262144 + m0 * 256 + ns + t * 16;
#pragma unroll
    for (int i = 0; i < 8; ++i) o[(i + 8 * half) * 256 + n] = (bf16)acc[t][i];
  }
}

// k_cheb: cheb = xg @ theta0 + adjxg @ theta1 (fused as K=512 accumulation).
// Each wave: one 16x64 strip (4 accumulators).
__global__ __launch_bounds__(128) void k_cheb(const bf16* __restrict__ xg,
                                              const bf16* __restrict__ adjxg,
                                              const bf16* __restrict__ thetaT,
                                              float* __restrict__ cheb) {
  const int wid = blockIdx.x * 4 + (threadIdx.x >> 5);   // 8192 waves
  const int b = wid >> 8, r = wid & 255;
  const int m0 = (r >> 2) * 16;        // 64 m-tiles
  const int ns = (r & 3) * 64;         // 4 n-strips over E=256
  const bf16* A0 = xg    + b * 262144 + m0 * 256;
  const bf16* A1 = adjxg + b * 262144 + m0 * 256;
  v8f acc[4] = {};
#pragma unroll
  for (int ks = 0; ks < 8; ++ks) {
    const v16bf a = frag_a_bf(A0 + ks * 32, 256);
#pragma unroll
    for (int t = 0; t < 4; ++t)
      acc[t] = wmma_bf16(a, frag_b_bf(thetaT + (ns + t * 16) * 256 + ks * 32, 256), acc[t]);
  }
#pragma unroll
  for (int ks = 0; ks < 8; ++ks) {
    const v16bf a = frag_a_bf(A1 + ks * 32, 256);
#pragma unroll
    for (int t = 0; t < 4; ++t)
      acc[t] = wmma_bf16(a, frag_b_bf(thetaT + 65536 + (ns + t * 16) * 256 + ks * 32, 256), acc[t]);
  }
  const int lane = threadIdx.x & 31, n = lane & 15, half = lane >> 4;
#pragma unroll
  for (int t = 0; t < 4; ++t) {
    float* o = cheb + b * 262144 + m0 * 256 + ns + t * 16;
#pragma unroll
    for (int i = 0; i < 8; ++i) o[(i + 8 * half) * 256 + n] = acc[t][i];
  }
}

// k_epi: out = relu(x_in + relu(cheb_reshape)); both flat-index identical.
__global__ void k_epi(const float* __restrict__ x_in, const float* __restrict__ cheb,
                      float* __restrict__ out) {
  const int i = blockIdx.x * blockDim.x + threadIdx.x;   // 8388608
  float g = cheb[i];
  g = g > 0.f ? g : 0.f;
  const float v = x_in[i] + g;
  out[i] = v > 0.f ? v : 0.f;
}

// ---------------------------------------------------------------------------
extern "C" void kernel_launch(void* const* d_in, const int* in_sizes, int n_in,
                              void* d_out, int out_size, void* d_ws, size_t ws_size,
                              hipStream_t stream) {
  const float* x         = (const float*)d_in[0];
  const float* se_w1     = (const float*)d_in[1];
  const float* se_w2     = (const float*)d_in[2];
  const float* ws_input  = (const float*)d_in[3];  // [E,L,1,1]
  const float* in_proj_w = (const float*)d_in[4];  // [3E,E]
  const float* in_proj_b = (const float*)d_in[5];  // [3E]
  const float* theta     = (const float*)d_in[6];  // [2,E,E]
  float* out = (float*)d_out;                       // [B*E*C] ++ [B*C*C]

  char* p = (char*)d_ws;
  float* y0    = (float*)p; p += 32768u * 4;          // [B,C]
  float* y1    = (float*)p; p += 32768u * 4;
  float* y2    = (float*)p; p += 32768u * 4;
  float* x_in  = (float*)p; p += 8388608u * 4;        // [B,E,C] f32
  float* cheb  = (float*)p; p += 8388608u * 4;        // [B,C,E] f32
  bf16*  wqkv  = (bf16*)p;  p += 196608u * 2;         // [3E,E]
  bf16*  thT   = (bf16*)p;  p += 131072u * 2;         // [2,E(n),E(k)]
  bf16*  xg    = (bf16*)p;  p += 8388608u * 2;        // [B,C,E]
  bf16*  xinb  = (bf16*)p;  p += 8388608u * 2;        // [B,E,C] (= xg^T)
  bf16*  qb    = (bf16*)p;  p += 8388608u * 2;        // [B,C,E] (pre-scaled)
  bf16*  kb    = (bf16*)p;  p += 8388608u * 2;        // [B,C,E]
  bf16*  adjxg = (bf16*)p;  p += 8388608u * 2;        // [B,C,E]

  float* badj = out + 8388608;                        // b_adj output region

  k_prep <<<dim3(1280),  dim3(256), 0, stream>>>(in_proj_w, theta, wqkv, thT);
  k_pool <<<dim3(128),   dim3(256), 0, stream>>>(x, y0);
  k_fc   <<<dim3(128),   dim3(256), 0, stream>>>(y0, se_w1, y1, 0);
  k_fc   <<<dim3(128),   dim3(256), 0, stream>>>(y1, se_w2, y2, 1);
  k_conv <<<dim3(32768), dim3(256), 0, stream>>>(x, ws_input, y2, x_in, xinb, xg);
  k_qkv  <<<dim3(4096),  dim3(128), 0, stream>>>(xg, wqkv, in_proj_b, qb, kb);
  k_attn <<<dim3(2048),  dim3(128), (2u * 16384u + 160u) * 4u, stream>>>(qb, kb, badj);
  k_adjxg<<<dim3(2048),  dim3(128), 0, stream>>>(badj, xinb, adjxg);
  k_cheb <<<dim3(2048),  dim3(128), 0, stream>>>(xg, adjxg, thT, cheb);
  k_epi  <<<dim3(32768), dim3(256), 0, stream>>>(x_in, cheb, out);
}
